// EgocentricHiveMind_11957188952410
// MI455X (gfx1250) — matrix operand order repaired and checked by reading, not compiled, not run on hardware
//
#include <hip/hip_runtime.h>
#include <hip/hip_bf16.h>
#include <stdint.h>

typedef __attribute__((ext_vector_type(16))) _Float16 v16h;
typedef __attribute__((ext_vector_type(8)))  _Float16 v8h;
typedef __attribute__((ext_vector_type(8)))  float    v8f;

namespace {

constexpr int Tn = 64, Bn = 32, An = 8, Nn = 16;
constexpr int S  = Tn * Bn * An;       // 16384
constexpr int Mn = 1 + 2 * Nn;         // 33 entities (self + 2*16)
constexpr int En = 64;
constexpr int DS = 290;
constexpr int K_SELF = 320;            // 290 padded to mult-of-32
constexpr int K_POOL = 384;            // 354 padded
constexpr int LI = 128, LH = 128;
constexpr int RB = Bn * An;            // 256 rows per LSTM step
constexpr int CHUNKS = 8;
constexpr int SC = S / CHUNKS;         // 2048 agent-steps per attention chunk
constexpr int RC = SC * Mn;            // 67584 entity rows per chunk

__device__ __forceinline__ float sigmf(float x) { return 1.0f / (1.0f + __expf(-x)); }

// ---- WMMA fragment loads (16x16x32 f16) -----------------------------------
// A (16x32, row M = lane&15): lanes 0-15 hold K {0..7,16..23}, lanes 16-31 hold
// K {8..15,24..31}; VGPRs 0-3 = first chunk, 4-7 = second (ISA 7.12.2).
__device__ __forceinline__ v16h load_a_frag(const _Float16* __restrict__ row, int k0, int hi) {
  v8h a0 = *reinterpret_cast<const v8h*>(row + k0 + 8 * hi);
  v8h a1 = *reinterpret_cast<const v8h*>(row + k0 + 16 + 8 * hi);
  return __builtin_shufflevector(a0, a1, 0,1,2,3,4,5,6,7,8,9,10,11,12,13,14,15);
}
// B (32x16, col N = lane&15): lanes 0-15 hold K {0..7,16..23}, lanes 16-31 hold K {8..15,24..31}.
__device__ __forceinline__ v16h load_b_frag(const _Float16* __restrict__ row, int k0, int hi) {
  v8h b0 = *reinterpret_cast<const v8h*>(row + k0 + 8 * hi);
  v8h b1 = *reinterpret_cast<const v8h*>(row + k0 + 16 + 8 * hi);
  return __builtin_shufflevector(b0, b1, 0,1,2,3,4,5,6,7,8,9,10,11,12,13,14,15);
}

// ---- Generic WMMA GEMM: C[r,c] = act( sum_k A[r,k]*W[c,k] + bias[c] ) ------
// One wave computes a 16x64 output strip (4 accumulators): one A fragment feeds
// 4 WMMAs per K-step. rows mult of 16, cols mult of 64, kd mult of 32.
__global__ __launch_bounds__(256) void gemm_wmma(
    const _Float16* __restrict__ A, int lda,
    const _Float16* __restrict__ W, int ldw,
    const float* __restrict__ bias,
    float* __restrict__ Cf, _Float16* __restrict__ Ch, int ldc,
    int rows, int cols, int kd, int relu) {
  int wave = threadIdx.x >> 5, lane = threadIdx.x & 31;
  int colStrips = cols >> 6;
  int totalStrips = (rows >> 4) * colStrips;
  int strip = blockIdx.x * (blockDim.x >> 5) + wave;
  if (strip >= totalStrips) return;  // wave-uniform, EXEC stays full
  int rt = strip / colStrips, cs = strip - rt * colStrips;
  int r0 = rt << 4, c0 = cs << 6;
  int hi = lane >> 4, lo = lane & 15;
  const _Float16* arow = A + (size_t)(r0 + lo) * lda;
  const _Float16* wrow = W + (size_t)(c0 + lo) * ldw;
  size_t wstep = (size_t)16 * ldw;
  v8f acc0 = {0.f,0.f,0.f,0.f,0.f,0.f,0.f,0.f};
  v8f acc1 = acc0, acc2 = acc0, acc3 = acc0;
  for (int k0 = 0; k0 < kd; k0 += 32) {
    v16h a  = load_a_frag(arow, k0, hi);
    v16h b0 = load_b_frag(wrow,             k0, hi);
    v16h b1 = load_b_frag(wrow + wstep,     k0, hi);
    v16h b2 = load_b_frag(wrow + 2 * wstep, k0, hi);
    v16h b3 = load_b_frag(wrow + 3 * wstep, k0, hi);
    acc0 = __builtin_amdgcn_wmma_f32_16x16x32_f16(false, a, false, b0, (short)0, acc0, false, false);
    acc1 = __builtin_amdgcn_wmma_f32_16x16x32_f16(false, a, false, b1, (short)0, acc1, false, false);
    acc2 = __builtin_amdgcn_wmma_f32_16x16x32_f16(false, a, false, b2, (short)0, acc2, false, false);
    acc3 = __builtin_amdgcn_wmma_f32_16x16x32_f16(false, a, false, b3, (short)0, acc3, false, false);
  }
  v8f accs[4] = {acc0, acc1, acc2, acc3};
#pragma unroll
  for (int j = 0; j < 4; ++j) {
    int n = c0 + 16 * j + lo;
    float bv = bias ? bias[n] : 0.f;
#pragma unroll
    for (int v = 0; v < 8; ++v) {
      int m = r0 + v + 8 * hi;           // C layout: VGPR v -> M = v + 8*hi
      float x = accs[j][v] + bv;
      if (relu) x = fmaxf(x, 0.f);
      size_t idx = (size_t)m * ldc + n;
      if (Cf) Cf[idx] = x;
      if (Ch) Ch[idx] = (_Float16)x;
    }
  }
}

// ---- sa GEMM with fused residual + mask + mean-pool epilogue ---------------
// res = zent + relu(oproj @ sa_w^T + b); zpool[s,:] += res * mask[s,m] / 33.
__global__ __launch_bounds__(256) void gemm_sa_pool(
    const _Float16* __restrict__ A,        // oproj chunk rows x 64
    const _Float16* __restrict__ W,        // sa_w f16 64x64
    const float* __restrict__ bias,
    const _Float16* __restrict__ zent,     // chunk rows x 64
    const float* __restrict__ maskb,       // full S*33
    float* __restrict__ zpool,             // full S x 64
    int rows, int rowBaseGlobal) {
  int wave = threadIdx.x >> 5, lane = threadIdx.x & 31;
  int totalStrips = rows >> 4;             // 64 cols = one strip
  int strip = blockIdx.x * (blockDim.x >> 5) + wave;
  if (strip >= totalStrips) return;
  int r0 = strip << 4;
  int hi = lane >> 4, lo = lane & 15;
  const _Float16* arow = A + (size_t)(r0 + lo) * 64;
  const _Float16* wrow = W + (size_t)lo * 64;
  v8f acc0 = {0.f,0.f,0.f,0.f,0.f,0.f,0.f,0.f};
  v8f acc1 = acc0, acc2 = acc0, acc3 = acc0;
  for (int k0 = 0; k0 < 64; k0 += 32) {
    v16h a  = load_a_frag(arow, k0, hi);
    v16h b0 = load_b_frag(wrow,            k0, hi);
    v16h b1 = load_b_frag(wrow + 16 * 64,  k0, hi);
    v16h b2 = load_b_frag(wrow + 32 * 64,  k0, hi);
    v16h b3 = load_b_frag(wrow + 48 * 64,  k0, hi);
    acc0 = __builtin_amdgcn_wmma_f32_16x16x32_f16(false, a, false, b0, (short)0, acc0, false, false);
    acc1 = __builtin_amdgcn_wmma_f32_16x16x32_f16(false, a, false, b1, (short)0, acc1, false, false);
    acc2 = __builtin_amdgcn_wmma_f32_16x16x32_f16(false, a, false, b2, (short)0, acc2, false, false);
    acc3 = __builtin_amdgcn_wmma_f32_16x16x32_f16(false, a, false, b3, (short)0, acc3, false, false);
  }
  v8f accs[4] = {acc0, acc1, acc2, acc3};
#pragma unroll
  for (int j = 0; j < 4; ++j) {
    int n = 16 * j + lo;
    float bv = bias[n];
#pragma unroll
    for (int v = 0; v < 8; ++v) {
      int m = r0 + v + 8 * hi;
      float x = fmaxf(accs[j][v] + bv, 0.f);
      float res = (float)zent[(size_t)m * 64 + n] + x;
      size_t g = (size_t)rowBaseGlobal + m;   // global entity row
      int s = (int)(g / (size_t)Mn);
      atomicAdd(&zpool[(size_t)s * En + n], res * maskb[g] * (1.0f / (float)Mn));
    }
  }
}

// ---- zero fill -------------------------------------------------------------
__global__ void zero_f32(float* __restrict__ p, int n) {
  int i = blockIdx.x * blockDim.x + threadIdx.x;
  if (i < n) p[i] = 0.f;
}

// ---- weight prep: f32 -> f16, offset/pad -----------------------------------
__global__ void prep_w(const float* __restrict__ src, _Float16* __restrict__ dst,
                       int rows, int srcK, int dstK, int srcOff, int copyLen) {
  int i = blockIdx.x * blockDim.x + threadIdx.x;
  if (i >= rows * dstK) return;
  int r = i / dstK, k = i - r * dstK;
  float v = (k < copyLen) ? src[(size_t)r * srcK + srcOff + k] : 0.f;
  dst[i] = (_Float16)v;
}
// self_w acts on concat([x,x]) -> fold: eff[e,k] = w[e,k] + w[e,290+k]
__global__ void prep_self(const float* __restrict__ sw, _Float16* __restrict__ dst) {
  int i = blockIdx.x * blockDim.x + threadIdx.x;
  if (i >= 64 * K_SELF) return;
  int e = i / K_SELF, k = i - e * K_SELF;
  float v = (k < DS) ? sw[(size_t)e * 580 + k] + sw[(size_t)e * 580 + DS + k] : 0.f;
  dst[i] = (_Float16)v;
}

// ---- conv1d(1->9,k=3) + x_self assembly ------------------------------------
__global__ void build_xself(const float* __restrict__ x_agent, const float* __restrict__ x_lidar,
                            const float* __restrict__ x_safe, const float* __restrict__ cw,
                            const float* __restrict__ cb,
                            _Float16* __restrict__ xself, _Float16* __restrict__ apool) {
  int i = blockIdx.x * blockDim.x + threadIdx.x;
  if (i >= S * K_POOL) return;
  int s = i / K_POOL, k = i - s * K_POOL;
  float v = 0.f;
  if (k < 16) {
    v = x_agent[(size_t)s * 16 + k];
  } else if (k < 286) {
    int j = k - 16, ch = j / 30, pos = j - ch * 30;
    const float* xl = x_lidar + (size_t)s * 32;
    v = cb[ch] + cw[ch * 3 + 0] * xl[pos] + cw[ch * 3 + 1] * xl[pos + 1] + cw[ch * 3 + 2] * xl[pos + 2];
    v = fmaxf(v, 0.f);
  } else if (k < DS) {
    v = x_safe[(size_t)s * 4 + (k - 286)];
  }
  apool[i] = (_Float16)v;          // [290,354) overwritten by pool-fill later
  if (k < K_SELF) xself[(size_t)s * K_SELF + k] = (_Float16)v;
}

// ---- entity mask (S x 33) --------------------------------------------------
__global__ void build_mask(const int* __restrict__ m0, const int* __restrict__ m1,
                           float* __restrict__ maskb) {
  int i = blockIdx.x * blockDim.x + threadIdx.x;
  if (i >= S * Mn) return;
  int s = i / Mn, m = i - s * Mn;
  float v = 1.f;
  if (m >= 1 && m <= Nn)       v = (float)m0[(size_t)s * Nn + m - 1];
  else if (m > Nn)             v = (float)m1[(size_t)s * Nn + m - 1 - Nn];
  maskb[i] = v;
}

// ---- entity encoders: z = relu(pre_self + W_ent @ feat + b) ----------------
__global__ void entity_enc(const float* __restrict__ xe0, const float* __restrict__ xe1,
                           const float* __restrict__ w0, const float* __restrict__ b0,
                           const float* __restrict__ w1, const float* __restrict__ b1,
                           const float* __restrict__ pre0, const float* __restrict__ pre1,
                           _Float16* __restrict__ zent) {
  int i = blockIdx.x * blockDim.x + threadIdx.x;
  if (i >= S * 2 * Nn * En) return;
  int e = i & 63, n = (i >> 6) & 15, ty = (i >> 10) & 1, s = i >> 11;
  float acc;
  if (ty == 0) {
    acc = pre0[(size_t)s * En + e] + b0[e];
    const float* f = xe0 + ((size_t)s * Nn + n) * 12;
    const float* w = w0 + (size_t)e * 302;  // ent0_w[:, :12]
#pragma unroll
    for (int j = 0; j < 12; ++j) acc += f[j] * w[j];
  } else {
    acc = pre1[(size_t)s * En + e] + b1[e];
    const float* f = xe1 + ((size_t)s * Nn + n) * 8;
    const float* w = w1 + (size_t)e * 298;  // ent1_w[:, :8]
#pragma unroll
    for (int j = 0; j < 8; ++j) acc += f[j] * w[j];
  }
  zent[((size_t)s * Mn + 1 + ty * Nn + n) * En + e] = (_Float16)fmaxf(acc, 0.f);
}

// ---- masked multi-head attention over 33 entities, one block per agent-step.
// qkv tile staged global->LDS with async-to-LDS DMA (ASYNCcnt), then VALU.
__global__ __launch_bounds__(256) void attn_kernel(const _Float16* __restrict__ qkv,
                                                   const float* __restrict__ maskb,
                                                   _Float16* __restrict__ o, int sBase) {
  __shared__ __align__(16) _Float16 sh[Mn * 192];
  __shared__ float shm[Mn];
  int tid = threadIdx.x;
  constexpr int BYTES = Mn * 192 * 2;   // 12672, mult of 16
  constexpr int NCH = BYTES / 16;       // 792 b128 chunks
  uint32_t ldsBase = (uint32_t)(uintptr_t)&sh[0];
  const char* gbase = (const char*)qkv + (size_t)blockIdx.x * BYTES;
  for (int i = tid; i < NCH; i += blockDim.x) {
    uint32_t la = ldsBase + (uint32_t)i * 16u;
    uint64_t ga = (uint64_t)(uintptr_t)(gbase + (size_t)i * 16);
    asm volatile("global_load_async_to_lds_b128 %0, %1, off"
                 :: "v"(la), "v"(ga) : "memory");
  }
  if (tid < Mn) shm[tid] = maskb[(size_t)(sBase + blockIdx.x) * Mn + tid];
  asm volatile("s_wait_asynccnt 0x0" ::: "memory");
  __syncthreads();
  int task = tid;               // 4 heads * 33 queries = 132 tasks
  if (task >= 4 * Mn) return;
  int h = task / Mn, m = task - h * Mn;
  float q[16];
#pragma unroll
  for (int d = 0; d < 16; ++d) q[d] = (float)sh[m * 192 + h * 16 + d];
  float sc[Mn];
  float mx = -3.0e38f;
#pragma unroll
  for (int j = 0; j < Mn; ++j) {
    float dp = 0.f;
#pragma unroll
    for (int d = 0; d < 16; ++d) dp += q[d] * (float)sh[j * 192 + 64 + h * 16 + d];
    dp *= 0.25f;                               // 1/sqrt(16)
    if (shm[j] <= 0.f) dp = -1.0e9f;
    sc[j] = dp;
    mx = fmaxf(mx, dp);
  }
  float se = 0.f;
#pragma unroll
  for (int j = 0; j < Mn; ++j) { float e = __expf(sc[j] - mx); sc[j] = e; se += e; }
  float inv = 1.0f / se;
#pragma unroll
  for (int d = 0; d < 16; ++d) {
    float acc = 0.f;
#pragma unroll
    for (int j = 0; j < Mn; ++j) acc += sc[j] * (float)sh[j * 192 + 128 + h * 16 + d];
    o[((size_t)blockIdx.x * Mn + m) * En + h * 16 + d] = (_Float16)(acc * inv);
  }
}

// ---- pool-fill: write z_pool (f16) into Apool[:, 290:354] ------------------
__global__ void pool_fill(const float* __restrict__ zpool, _Float16* __restrict__ apool) {
  int i = blockIdx.x * blockDim.x + threadIdx.x;
  if (i >= S * En) return;
  int s = i >> 6, e = i & 63;
  apool[(size_t)s * K_POOL + DS + e] = (_Float16)zpool[i];
}

// ---- LSTM: pre-GEMM done-mask on h ----------------------------------------
__global__ void lstm_mask(const _Float16* __restrict__ h, const int* __restrict__ done,
                          _Float16* __restrict__ hm, int t) {
  int i = blockIdx.x * blockDim.x + threadIdx.x;
  if (i >= RB * LH) return;
  int r = i >> 7;
  float keep = done[t * Bn + (r >> 3)] ? 0.f : 1.f;
  hm[i] = (_Float16)((float)h[i] * keep);
}

// ---- LSTM cell (gate order i,f,g,o) ----------------------------------------
__global__ void lstm_cell(const float* __restrict__ Gx, const float* __restrict__ Gh,
                          const float* __restrict__ bih, const float* __restrict__ bhh,
                          const int* __restrict__ done, float* __restrict__ c,
                          _Float16* __restrict__ h, float* __restrict__ zlstm, int t) {
  int i = blockIdx.x * blockDim.x + threadIdx.x;
  if (i >= RB * LH) return;
  int r = i >> 7, j = i & 127;
  float keep = done[t * Bn + (r >> 3)] ? 0.f : 1.f;
  const float* gx = Gx + ((size_t)t * RB + r) * 512;
  const float* gh = Gh + (size_t)r * 512;
  float gi = gx[j]       + gh[j]       + bih[j]       + bhh[j];
  float gf = gx[128 + j] + gh[128 + j] + bih[128 + j] + bhh[128 + j];
  float gg = gx[256 + j] + gh[256 + j] + bih[256 + j] + bhh[256 + j];
  float go = gx[384 + j] + gh[384 + j] + bih[384 + j] + bhh[384 + j];
  float cn = sigmf(gf) * (c[i] * keep) + sigmf(gi) * tanhf(gg);
  float hn = sigmf(go) * tanhf(cn);
  c[i] = cn;
  h[i] = (_Float16)hn;
  zlstm[(size_t)t * RB * LH + i] = hn;
}

// ---- categorical heads + outputs -------------------------------------------
__device__ void head_eval(const float* __restrict__ z, const float* __restrict__ w,
                          const float* __restrict__ b, int act, float& lp, float& ent) {
  float l[5];
#pragma unroll
  for (int cc = 0; cc < 5; ++cc) {
    float s = b[cc];
    const float* wr = w + cc * LH;
    for (int k = 0; k < LH; ++k) s += z[k] * wr[k];
    l[cc] = s;
  }
  float mx = l[0];
#pragma unroll
  for (int cc = 1; cc < 5; ++cc) mx = fmaxf(mx, l[cc]);
  float se = 0.f;
#pragma unroll
  for (int cc = 0; cc < 5; ++cc) se += __expf(l[cc] - mx);
  float logZ = mx + __logf(se);
  float e = 0.f;
#pragma unroll
  for (int cc = 0; cc < 5; ++cc) { float p = __expf(l[cc] - logZ); e -= p * (l[cc] - logZ); }
  ent = e;
  lp = l[act] - logZ;
}

__global__ void heads_kernel(const float* __restrict__ zlstm,
                             const float* __restrict__ h0w, const float* __restrict__ h0b,
                             const float* __restrict__ h1w, const float* __restrict__ h1b,
                             const int* __restrict__ actions, float* __restrict__ out) {
  int i = blockIdx.x * blockDim.x + threadIdx.x;
  if (i >= S) return;
  const float* z = zlstm + (size_t)i * LH;
  int a0 = actions[(size_t)i * 2 + 0], a1 = actions[(size_t)i * 2 + 1];
  float lp0, lp1, e0, e1;
  head_eval(z, h0w, h0b, a0, lp0, e0);
  head_eval(z, h1w, h1b, a1, lp1, e1);
  out[(size_t)i * 2 + 0] = (float)a0;             // actions (as float), T*B*A*2
  out[(size_t)i * 2 + 1] = (float)a1;
  out[2 * S + i] = lp0 * lp1;                     // logprobs
  out[3 * S + (size_t)i * 2 + 0] = e0;            // entropies (T,B,A,2)
  out[3 * S + (size_t)i * 2 + 1] = e1;
}

}  // namespace

extern "C" void kernel_launch(void* const* d_in, const int* in_sizes, int n_in,
                              void* d_out, int out_size, void* d_ws, size_t ws_size,
                              hipStream_t stream) {
  // inputs (setup_inputs dict order)
  const float* x_agent  = (const float*)d_in[0];
  const float* x_lidar  = (const float*)d_in[1];
  const float* x_safe   = (const float*)d_in[2];
  const float* x_ent0   = (const float*)d_in[3];
  const float* x_ent1   = (const float*)d_in[4];
  const int*   mask0    = (const int*)d_in[5];
  const int*   mask1    = (const int*)d_in[6];
  const int*   done     = (const int*)d_in[7];
  const int*   actions  = (const int*)d_in[8];
  const float* conv_w   = (const float*)d_in[9];
  const float* conv_b   = (const float*)d_in[10];
  const float* self_w   = (const float*)d_in[11];
  const float* self_b   = (const float*)d_in[12];
  const float* ent0_w   = (const float*)d_in[13];
  const float* ent0_b   = (const float*)d_in[14];
  const float* ent1_w   = (const float*)d_in[15];
  const float* ent1_b   = (const float*)d_in[16];
  const float* inproj_w = (const float*)d_in[17];
  const float* inproj_b = (const float*)d_in[18];
  const float* outproj_w= (const float*)d_in[19];
  const float* outproj_b= (const float*)d_in[20];
  const float* sa_w     = (const float*)d_in[21];
  const float* sa_b     = (const float*)d_in[22];
  const float* pool_w   = (const float*)d_in[23];
  const float* pool_b   = (const float*)d_in[24];
  const float* w_ih     = (const float*)d_in[25];
  const float* w_hh     = (const float*)d_in[26];
  const float* b_ih     = (const float*)d_in[27];
  const float* b_hh     = (const float*)d_in[28];
  const float* h0w      = (const float*)d_in[29];
  const float* h0b      = (const float*)d_in[30];
  const float* h1w      = (const float*)d_in[31];
  const float* h1b      = (const float*)d_in[32];

  // workspace bump allocator (256B aligned)
  char* base = (char*)d_ws;
  size_t off = 0;
  auto allocB = [&](size_t bytes) -> char* {
    char* p = base + off;
    off = (off + bytes + 255) & ~(size_t)255;
    return p;
  };
  auto allocH = [&](size_t n) { return (_Float16*)allocB(n * 2); };
  auto allocF = [&](size_t n) { return (float*)allocB(n * 4); };

  _Float16* xself  = allocH((size_t)S * K_SELF);
  _Float16* apool  = allocH((size_t)S * K_POOL);
  _Float16* zent   = allocH((size_t)S * Mn * En);
  float*    pre0   = allocF((size_t)S * En);
  float*    pre1   = allocF((size_t)S * En);
  float*    maskb  = allocF((size_t)S * Mn);
  _Float16* qkv    = allocH((size_t)RC * 192);
  _Float16* o_h    = allocH((size_t)RC * En);
  _Float16* oproj  = allocH((size_t)RC * En);
  float*    zpool  = allocF((size_t)S * En);
  _Float16* zin    = allocH((size_t)S * LI);
  float*    Gx     = allocF((size_t)S * 512);
  float*    Gh     = allocF((size_t)RB * 512);
  _Float16* h_h    = allocH((size_t)RB * LH);
  _Float16* h_m    = allocH((size_t)RB * LH);
  float*    c_st   = allocF((size_t)RB * LH);
  float*    zlstm  = allocF((size_t)S * LH);
  _Float16* wself  = allocH(64 * K_SELF);
  _Float16* wb0    = allocH(64 * K_SELF);
  _Float16* wb1    = allocH(64 * K_SELF);
  _Float16* winp   = allocH(192 * 64);
  _Float16* woutp  = allocH(64 * 64);
  _Float16* wsa    = allocH(64 * 64);
  _Float16* wpool  = allocH(128 * K_POOL);
  _Float16* wih    = allocH(512 * 128);
  _Float16* whh    = allocH(512 * 128);

  auto gemm = [&](const _Float16* A, int lda, const _Float16* W, int ldw, const float* bias,
                  float* Cf, _Float16* Ch, int ldc, int rows, int cols, int kd, int relu) {
    int strips = (rows >> 4) * (cols >> 6);
    int blocks = (strips + 7) / 8;
    gemm_wmma<<<blocks, 256, 0, stream>>>(A, lda, W, ldw, bias, Cf, Ch, ldc, rows, cols, kd, relu);
  };
  auto grid = [](int n) { return (n + 255) / 256; };

  // ---- weight prep (cheap, recomputed each call: deterministic) ----
  prep_self<<<grid(64 * K_SELF), 256, 0, stream>>>(self_w, wself);
  prep_w<<<grid(64 * K_SELF), 256, 0, stream>>>(ent0_w, wb0, 64, 302, K_SELF, 12, DS);
  prep_w<<<grid(64 * K_SELF), 256, 0, stream>>>(ent1_w, wb1, 64, 298, K_SELF, 8, DS);
  prep_w<<<grid(192 * 64), 256, 0, stream>>>(inproj_w, winp, 192, 64, 64, 0, 64);
  prep_w<<<grid(64 * 64), 256, 0, stream>>>(outproj_w, woutp, 64, 64, 64, 0, 64);
  prep_w<<<grid(64 * 64), 256, 0, stream>>>(sa_w, wsa, 64, 64, 64, 0, 64);
  prep_w<<<grid(128 * K_POOL), 256, 0, stream>>>(pool_w, wpool, 128, 354, K_POOL, 0, 354);
  prep_w<<<grid(512 * 128), 256, 0, stream>>>(w_ih, wih, 512, 128, 128, 0, 128);
  prep_w<<<grid(512 * 128), 256, 0, stream>>>(w_hh, whh, 512, 128, 128, 0, 128);

  // ---- per-call state init ----
  zero_f32<<<grid(S * En), 256, 0, stream>>>(zpool, S * En);
  zero_f32<<<grid(RB * LH), 256, 0, stream>>>(c_st, RB * LH);
  zero_f32<<<grid(RB * LH / 2), 256, 0, stream>>>((float*)h_h, RB * LH / 2);

  // ---- conv + x_self, masks ----
  build_xself<<<grid(S * K_POOL), 256, 0, stream>>>(x_agent, x_lidar, x_safe, conv_w, conv_b,
                                                    xself, apool);
  build_mask<<<grid(S * Mn), 256, 0, stream>>>(mask0, mask1, maskb);

  // ---- z_self (into zent row m=0, ldc spans all 33 entities) ----
  gemm(xself, K_SELF, wself, K_SELF, self_b, nullptr, zent, Mn * En, S, En, K_SELF, 1);
  // ---- broadcast self-parts of entity encoders ----
  gemm(xself, K_SELF, wb0, K_SELF, nullptr, pre0, nullptr, En, S, En, K_SELF, 0);
  gemm(xself, K_SELF, wb1, K_SELF, nullptr, pre1, nullptr, En, S, En, K_SELF, 0);
  // ---- entity-feature part (K=12/8) + relu, write zent m=1..32 ----
  entity_enc<<<grid(S * 2 * Nn * En), 256, 0, stream>>>(x_ent0, x_ent1, ent0_w, ent0_b,
                                                        ent1_w, ent1_b, pre0, pre1, zent);

  // ---- attention pipeline, chunked over T to bound workspace ----
  for (int cc = 0; cc < CHUNKS; ++cc) {
    const _Float16* zc = zent + (size_t)cc * RC * En;
    gemm(zc, En, winp, 64, inproj_b, nullptr, qkv, 192, RC, 192, 64, 0);
    attn_kernel<<<SC, 256, 0, stream>>>(qkv, maskb, o_h, cc * SC);
    gemm(o_h, En, woutp, 64, outproj_b, nullptr, oproj, En, RC, En, 64, 0);
    int strips = RC >> 4;
    gemm_sa_pool<<<(strips + 7) / 8, 256, 0, stream>>>(oproj, wsa, sa_b, zc, maskb, zpool,
                                                       RC, cc * RC);
  }

  // ---- pool MLP ----
  pool_fill<<<grid(S * En), 256, 0, stream>>>(zpool, apool);
  gemm(apool, K_POOL, wpool, K_POOL, pool_b, nullptr, zin, LI, S, LI, K_POOL, 1);

  // ---- LSTM: input-gate GEMM for all timesteps at once ----
  gemm(zin, LI, wih, 128, nullptr, Gx, nullptr, 512, S, 512, 128, 0);
  // ---- sequential scan: only h @ w_hh^T per step ----
  for (int t = 0; t < Tn; ++t) {
    lstm_mask<<<grid(RB * LH), 256, 0, stream>>>(h_h, done, h_m, t);
    gemm(h_m, LH, whh, 128, nullptr, Gh, nullptr, 512, RB, 512, 128, 0);
    lstm_cell<<<grid(RB * LH), 256, 0, stream>>>(Gx, Gh, b_ih, b_hh, done, c_st, h_h, zlstm, t);
  }

  // ---- heads + output assembly ----
  heads_kernel<<<grid(S), 256, 0, stream>>>(zlstm, h0w, h0b, h1w, h1b, actions, (float*)d_out);
}
